// EfficientCrossAttention_43026982371636
// MI455X (gfx1250) — matrix operand order repaired and checked by reading, not compiled, not run on hardware
//
#include <hip/hip_runtime.h>
#include <math.h>

typedef __bf16 bf16;
typedef __attribute__((ext_vector_type(16))) __bf16 v16bf;
typedef __attribute__((ext_vector_type(8)))  float  v8f;
typedef int v4i __attribute__((vector_size(16)));

#define Bsz 16
#define Tsz 2048
#define Dsz 1024
#define Hsz 16
#define ROWS (Bsz * Tsz)   // 32768

#if defined(__has_builtin)
#if __has_builtin(__builtin_amdgcn_global_load_async_to_lds_b128)
#define HAVE_ASYNC_LDS 1
#endif
#endif
#ifndef HAVE_ASYNC_LDS
#define HAVE_ASYNC_LDS 0
#endif

union Frag32B {
    uint4 u[2];
    v16bf v;
    __device__ Frag32B() {}
};

// 16-byte global->LDS copy: async on CDNA5 (ASYNCcnt path, no VGPR transit),
// VGPR round-trip fallback otherwise.
__device__ __forceinline__ void copy16_g2lds(const void* g, void* l)
{
#if HAVE_ASYNC_LDS
    __builtin_amdgcn_global_load_async_to_lds_b128(
        (__attribute__((address_space(1))) v4i*)(void*)g,
        (__attribute__((address_space(3))) v4i*)l, 0, 0);
#else
    *(uint4*)l = *(const uint4*)g;
#endif
}

__device__ __forceinline__ void wait_async_copies()
{
#if HAVE_ASYNC_LDS
#if defined(__has_builtin) && __has_builtin(__builtin_amdgcn_s_wait_asynccnt)
    __builtin_amdgcn_s_wait_asynccnt(0);
#else
    asm volatile("s_wait_asynccnt 0x0" ::: "memory");
#endif
#endif
}

// ---------------------------------------------------------------------------
// Weight conversion: f32 row-major [K][N] -> bf16 transposed [N][K]
// ---------------------------------------------------------------------------
__global__ __launch_bounds__(256) void convert_weights(
    const float* __restrict__ qkv_w,   // [1024][3072]
    const float* __restrict__ out_w,   // [1024][1024]
    bf16* __restrict__ qkv_wT,         // [3072][1024]
    bf16* __restrict__ out_wT)         // [1024][1024]
{
    size_t id = (size_t)blockIdx.x * 256 + threadIdx.x;
    const size_t n1 = 3072ull * 1024ull;
    if (id < n1) {
        size_t n = id / 1024, k = id % 1024;
        qkv_wT[id] = (bf16)qkv_w[k * 3072 + n];
    } else {
        size_t i2 = id - n1;
        if (i2 < 1024ull * 1024ull) {
            size_t n = i2 / 1024, k = i2 % 1024;
            out_wT[i2] = (bf16)out_w[k * 1024 + n];
        }
    }
}

// ---------------------------------------------------------------------------
// Block-wide sum reduction (256 threads = 8 waves, wave32)
// ---------------------------------------------------------------------------
__device__ __forceinline__ float block_sum(float v, float* red8)
{
    #pragma unroll
    for (int o = 16; o > 0; o >>= 1) v += __shfl_xor(v, o, 32);
    const int tid = threadIdx.x;
    if ((tid & 31) == 0) red8[tid >> 5] = v;
    __syncthreads();
    float tot = 0.f;
    #pragma unroll
    for (int i = 0; i < 8; ++i) tot += red8[i];
    __syncthreads();
    return tot;
}

// ---------------------------------------------------------------------------
// LayerNorm rows of x -> bf16 (one 256-thread block per row of 1024)
// ---------------------------------------------------------------------------
__global__ __launch_bounds__(256) void ln_rows_bf16(
    const float* __restrict__ x, const float* __restrict__ w,
    const float* __restrict__ b, bf16* __restrict__ out)
{
    __shared__ float red8[8];
    const int row = blockIdx.x;
    const int tid = threadIdx.x;
    const float* xr = x + (size_t)row * Dsz;
    float v[4], s = 0.f, s2 = 0.f;
    #pragma unroll
    for (int j = 0; j < 4; ++j) {
        v[j] = xr[tid + 256 * j];
        s += v[j]; s2 += v[j] * v[j];
    }
    s  = block_sum(s,  red8);
    s2 = block_sum(s2, red8);
    const float mu = s * (1.f / Dsz);
    const float rs = rsqrtf(s2 * (1.f / Dsz) - mu * mu + 1e-5f);
    #pragma unroll
    for (int j = 0; j < 4; ++j) {
        const int c = tid + 256 * j;
        out[(size_t)row * Dsz + c] = (bf16)((v[j] - mu) * rs * w[c] + b[c]);
    }
}

// ---------------------------------------------------------------------------
// bf16 WMMA GEMM:  C[M,N] = A[M,K] * WT[N,K]^T + bias (+ resid)
// Block tile 64(M) x 256(N), 8 waves in 2(M) x 4(N) grid,
// each wave: 2x4 tiles of v_wmma_f32_16x16x32_bf16.
// A staged through LDS via async global->LDS copies (ASYNCcnt path);
// B fragments loaded straight from L2-resident transposed weights.
// ---------------------------------------------------------------------------
template <bool RES>
__global__ __launch_bounds__(256) void gemm_bf16_wmma(
    const bf16* __restrict__ A, const bf16* __restrict__ WT,
    const float* __restrict__ bias, const float* __restrict__ resid,
    float* __restrict__ C, int N, int K)
{
    __shared__ bf16 smA[64 * 40];          // padded stride: 40 elems = 80 B
    const int tid  = threadIdx.x;
    const int lane = tid & 31;
    const int wv   = tid >> 5;
    const int wm   = wv & 1;               // 0..1
    const int wn   = wv >> 1;              // 0..3
    const int half = lane >> 4;            // 0..1
    const int l16  = lane & 15;
    const int mBase = blockIdx.x * 64;
    const int nBase = blockIdx.y * 256;

    const v8f vzero = {0.f, 0.f, 0.f, 0.f, 0.f, 0.f, 0.f, 0.f};
    v8f acc[2][4];
    #pragma unroll
    for (int i = 0; i < 2; ++i)
        #pragma unroll
        for (int s = 0; s < 4; ++s) acc[i][s] = vzero;

    // per-thread A staging coords: 8 bf16 (16B) each
    const int ar = tid >> 2;               // 0..63
    const int ac = (tid & 3) * 8;          // 0,8,16,24
    const bf16* agsrc = A + (size_t)(mBase + ar) * K + ac;
    bf16* aldst = &smA[ar * 40 + ac];

    for (int k0 = 0; k0 < K; k0 += 32) {
        // stage A tile 64x32 into LDS (async on CDNA5)
        copy16_g2lds(agsrc + k0, aldst);
        __builtin_prefetch(agsrc + k0 + 32, 0, 3);   // next K-step of A
        wait_async_copies();
        __syncthreads();

        // B fragments straight from global (transposed weights, K-contiguous)
        Frag32B bf[4];
        #pragma unroll
        for (int s = 0; s < 4; ++s) {
            const bf16* bp = WT + (size_t)(nBase + wn * 64 + s * 16 + l16) * K
                                + k0 + half * 16;
            bf[s].u[0] = *(const uint4*)bp;
            bf[s].u[1] = *(const uint4*)(bp + 8);
            __builtin_prefetch(bp + 32, 0, 3);       // next K-step of B
        }
        // A fragments from LDS (16-bit A 16x32 layout)
        Frag32B af[2];
        #pragma unroll
        for (int i = 0; i < 2; ++i) {
            const bf16* ap = &smA[(wm * 32 + i * 16 + l16) * 40 + half * 16];
            af[i].u[0] = *(const uint4*)ap;
            af[i].u[1] = *(const uint4*)(ap + 8);
        }
        #pragma unroll
        for (int i = 0; i < 2; ++i)
            #pragma unroll
            for (int s = 0; s < 4; ++s)
                acc[i][s] = __builtin_amdgcn_wmma_f32_16x16x32_bf16(
                    false, af[i].v, false, bf[s].v, (short)0, acc[i][s],
                    false, false);
        __syncthreads();
    }

    // epilogue: C 16x16 f32 layout -> row = half*8 + r, col = l16
    #pragma unroll
    for (int i = 0; i < 2; ++i)
        #pragma unroll
        for (int s = 0; s < 4; ++s)
            #pragma unroll
            for (int r = 0; r < 8; ++r) {
                const int m = mBase + wm * 32 + i * 16 + half * 8 + r;
                const int n = nBase + wn * 64 + s * 16 + l16;
                float val = acc[i][s][r] + bias[n];
                if (RES) val += resid[(size_t)m * N + n];
                C[(size_t)m * N + n] = val;
            }
}

// ---------------------------------------------------------------------------
// emb_out = silu(emb) @ emb_w + emb_b      [16][2048] @ [2048][2048]
// ---------------------------------------------------------------------------
__global__ __launch_bounds__(256) void emb_proj(
    const float* __restrict__ emb, const float* __restrict__ ew,
    const float* __restrict__ eb, float* __restrict__ eo)
{
    __shared__ float se[2048];
    const int b  = blockIdx.x >> 3;
    const int jb = (blockIdx.x & 7) * 256;
    const int tid = threadIdx.x;
    #pragma unroll
    for (int u = 0; u < 8; ++u) {
        const int k = tid + u * 256;
        const float e = emb[(size_t)b * 2048 + k];
        se[k] = e / (1.f + __expf(-e));
    }
    __syncthreads();
    const int j = jb + tid;
    float a = eb[j];
    for (int k = 0; k < 2048; ++k) a += se[k] * ew[(size_t)k * 2048 + j];
    eo[(size_t)b * 2048 + j] = a;
}

// ---------------------------------------------------------------------------
// k-softmax over T (with mask) and att[b,h] = k_sm^T @ (v*mask)  -> [64][64]
// one block per (b,h); register-tiled outer product, 16 accum/thread
// ---------------------------------------------------------------------------
__global__ __launch_bounds__(256) void kv_att(
    const float* __restrict__ qkv, const float* __restrict__ mask,
    float* __restrict__ att_out)
{
    const int bh = blockIdx.x;
    const int b = bh >> 4, h = bh & 15;
    __shared__ float colmax[64];
    __shared__ float red[256];
    __shared__ float ck[8 * 64];
    __shared__ float cv[8 * 64];
    const int tid = threadIdx.x;

    // pass 1: per-column (d) max over T
    {
        const int d = tid >> 2, tq = tid & 3;
        float mx = -1e30f;
        for (int t = tq; t < Tsz; t += 4) {
            const size_t r = (size_t)b * Tsz + t;
            const float m = mask[r];
            const float kv = qkv[r * 3072 + 1024 + h * 64 + d]
                             + (1.f - m) * (-1000000.0f);
            mx = fmaxf(mx, kv);
        }
        red[tid] = mx;
        __syncthreads();
        if (tq == 0)
            colmax[d] = fmaxf(fmaxf(red[tid], red[tid + 1]),
                              fmaxf(red[tid + 2], red[tid + 3]));
        __syncthreads();
    }

    // pass 2: accumulate e(k) outer v, plus column sums
    const int d2 = tid & 63;
    const int lb = (tid >> 6) * 16;
    const float cm = colmax[d2];
    float acc[16];
    #pragma unroll
    for (int u = 0; u < 16; ++u) acc[u] = 0.f;
    float ssum = 0.f;

    for (int t0 = 0; t0 < Tsz; t0 += 8) {
        __syncthreads();
        #pragma unroll
        for (int u = 0; u < 2; ++u) {
            const int id = tid * 2 + u;
            const int tt = id >> 6, c = id & 63;
            const size_t r = (size_t)b * Tsz + t0 + tt;
            const float m = mask[r];
            ck[id] = qkv[r * 3072 + 1024 + h * 64 + c]
                     + (1.f - m) * (-1000000.0f);
            cv[id] = qkv[r * 3072 + 2048 + h * 64 + c] * m;
        }
        __syncthreads();
        #pragma unroll
        for (int tt = 0; tt < 8; ++tt) {
            const float e = __expf(ck[tt * 64 + d2] - cm);
            ssum += e;
            const float* vr = &cv[tt * 64 + lb];
            #pragma unroll
            for (int u = 0; u < 16; ++u) acc[u] += e * vr[u];
        }
    }
    const float inv = 1.f / ssum;
    float* op = att_out + ((size_t)bh * 64 + d2) * 64 + lb;
    #pragma unroll
    for (int u = 0; u < 16; ++u) op[u] = acc[u] * inv;
}

// ---------------------------------------------------------------------------
// Fused: q-softmax (per head, dh=64) -> y = q_sm @ att -> LN(y) ->
//        *(1+scale)+shift -> silu -> bf16.  One block per (b,t) row.
// ---------------------------------------------------------------------------
__global__ __launch_bounds__(256) void attn_y_mod(
    const float* __restrict__ qkv, const float* __restrict__ att,
    const float* __restrict__ emb_out, const float* __restrict__ sn_w,
    const float* __restrict__ sn_b, bf16* __restrict__ h_out)
{
    __shared__ float qs[1024];
    __shared__ float red8[8];
    const int row = blockIdx.x;
    const int b = row >> 11;                 // T = 2048
    const int tid = threadIdx.x;

    // per-head softmax over 64 (16 threads per head, 4 elems each)
    {
        const int hh = tid >> 4, j16 = tid & 15;
        const float* qrow = qkv + (size_t)row * 3072 + hh * 64 + j16 * 4;
        float qv[4];
        float mx = -1e30f;
        #pragma unroll
        for (int u = 0; u < 4; ++u) { qv[u] = qrow[u]; mx = fmaxf(mx, qv[u]); }
        #pragma unroll
        for (int o = 8; o > 0; o >>= 1) mx = fmaxf(mx, __shfl_xor(mx, o, 16));
        float se = 0.f;
        #pragma unroll
        for (int u = 0; u < 4; ++u) { qv[u] = __expf(qv[u] - mx); se += qv[u]; }
        #pragma unroll
        for (int o = 8; o > 0; o >>= 1) se += __shfl_xor(se, o, 16);
        const float inv = 1.f / se;
        #pragma unroll
        for (int u = 0; u < 4; ++u) qs[hh * 64 + j16 * 4 + u] = qv[u] * inv;
    }
    __syncthreads();

    // y: 4 consecutive outputs per thread (same head)
    const int c0 = tid * 4;
    const int h2 = c0 >> 6, l0 = c0 & 63;
    const float* ap = att + ((size_t)(b * Hsz + h2) * 64) * 64 + l0;
    const float* qh = &qs[h2 * 64];
    float y[4] = {0.f, 0.f, 0.f, 0.f};
    #pragma unroll 8
    for (int d = 0; d < 64; ++d) {
        const float4 a4 = *(const float4*)(ap + d * 64);
        const float qd = qh[d];
        y[0] += qd * a4.x; y[1] += qd * a4.y;
        y[2] += qd * a4.z; y[3] += qd * a4.w;
    }

    // LayerNorm over the full 1024-wide row
    float s = y[0] + y[1] + y[2] + y[3];
    float s2 = y[0]*y[0] + y[1]*y[1] + y[2]*y[2] + y[3]*y[3];
    s  = block_sum(s,  red8);
    s2 = block_sum(s2, red8);
    const float mu = s * (1.f / Dsz);
    const float rs = rsqrtf(s2 * (1.f / Dsz) - mu * mu + 1e-5f);

    const float* sc = emb_out + (size_t)b * 2048;
    #pragma unroll
    for (int u = 0; u < 4; ++u) {
        const int c = c0 + u;
        const float hn = (y[u] - mu) * rs * sn_w[c] + sn_b[c];
        const float md = hn * (1.f + sc[c]) + sc[Dsz + c];
        const float sl = md / (1.f + __expf(-md));
        h_out[(size_t)row * Dsz + c] = (bf16)sl;
    }
}

// ---------------------------------------------------------------------------
extern "C" void kernel_launch(void* const* d_in, const int* in_sizes, int n_in,
                              void* d_out, int out_size, void* d_ws, size_t ws_size,
                              hipStream_t stream)
{
    const float* x       = (const float*)d_in[0];
    const float* src_mask= (const float*)d_in[1];
    const float* emb     = (const float*)d_in[2];
    const float* ln_w    = (const float*)d_in[3];
    const float* ln_b    = (const float*)d_in[4];
    const float* qkv_w   = (const float*)d_in[5];
    const float* qkv_b   = (const float*)d_in[6];
    const float* emb_w   = (const float*)d_in[7];
    const float* emb_b   = (const float*)d_in[8];
    const float* sn_w    = (const float*)d_in[9];
    const float* sn_b    = (const float*)d_in[10];
    const float* out_w   = (const float*)d_in[11];
    const float* out_b   = (const float*)d_in[12];
    float* out = (float*)d_out;

    size_t off = 0;
    auto carve = [&](size_t bytes) -> void* {
        void* p = (char*)d_ws + off;
        off += (bytes + 255) & ~(size_t)255;
        return p;
    };
    bf16*  xn      = (bf16*) carve((size_t)ROWS * Dsz * sizeof(bf16));
    bf16*  hin     = (bf16*) carve((size_t)ROWS * Dsz * sizeof(bf16));
    bf16*  qkv_wT  = (bf16*) carve((size_t)3 * Dsz * Dsz * sizeof(bf16));
    bf16*  out_wT  = (bf16*) carve((size_t)Dsz * Dsz * sizeof(bf16));
    float* qkvb    = (float*)carve((size_t)ROWS * 3 * Dsz * sizeof(float));
    float* attb    = (float*)carve((size_t)Bsz * Hsz * 64 * 64 * sizeof(float));
    float* embo    = (float*)carve((size_t)Bsz * 2 * Dsz * sizeof(float));
    (void)ws_size; (void)in_sizes; (void)n_in; (void)out_size;

    // 1. bf16 transposed weights
    convert_weights<<<(3072 * 1024 + 1024 * 1024 + 255) / 256, 256, 0, stream>>>(
        qkv_w, out_w, qkv_wT, out_wT);
    // 2. LayerNorm(x) -> bf16
    ln_rows_bf16<<<ROWS, 256, 0, stream>>>(x, ln_w, ln_b, xn);
    // 3. QKV GEMM (WMMA): [32768,1024]x[1024,3072]
    gemm_bf16_wmma<false><<<dim3(ROWS / 64, 3072 / 256), 256, 0, stream>>>(
        xn, qkv_wT, qkv_b, nullptr, qkvb, 3072, 1024);
    // 4. emb projection
    emb_proj<<<Bsz * 8, 256, 0, stream>>>(emb, emb_w, emb_b, embo);
    // 5. k-softmax + K^T V
    kv_att<<<Bsz * Hsz, 256, 0, stream>>>(qkvb, src_mask, attb);
    // 6. q-softmax + y + LN + modulation + silu -> bf16
    attn_y_mod<<<ROWS, 256, 0, stream>>>(qkvb, attb, embo, sn_w, sn_b, hin);
    // 7. out GEMM (WMMA) + bias + residual
    gemm_bf16_wmma<true><<<dim3(ROWS / 64, 1024 / 256), 256, 0, stream>>>(
        hin, out_wT, out_b, x, out, 1024, 1024);
}
